// MS_Attention_RepConv_qkv_id_23167053594839
// MI455X (gfx1250) — matrix operand order repaired and checked by reading, not compiled, not run on hardware
//
#include <hip/hip_runtime.h>
#include <hip/hip_bf16.h>

// ---------------------------------------------------------------------------
// MS_Attention_RepConv_qkv on MI455X (gfx1250, wave32, WMMA, async-LDS DMA).
//
// All matrix work runs through V_WMMA_F32_16X16X32_F16. Intermediates are
// fp16 (binary spikes stored as fp16 0/1 so they feed WMMA directly); BN
// chains fold to per-channel (scale,bias) in GEMM epilogues; LIF thresholds
// fuse there too. Workload is HBM bandwidth bound (~1.3 GB vs 120 GFLOP).
// The conv1x1 GEMMs stage tiles with GLOBAL_LOAD_ASYNC_TO_LDS_B128 (ASYNCcnt)
// into double-buffered LDS so the next K-slab's DMA overlaps current WMMAs.
// ---------------------------------------------------------------------------

typedef __attribute__((ext_vector_type(16))) _Float16 v16h;
typedef __attribute__((ext_vector_type(8)))  _Float16 v8h;
typedef __attribute__((ext_vector_type(8)))  float    v8f;

#define BATCH 16
#define CHN   256
#define HH    80
#define WW    80
#define HW    (HH * WW)                  // 6400
#define NELEM ((size_t)BATCH * CHN * HW) // 26,214,400

// ---------------------------------------------------------------------------
// CDNA5 async global->LDS copy (tracked by ASYNCcnt).
// GVS mode: mem_addr = SADDR(64) + VADDR(32) + IOFFSET ; VDST = LDS byte addr.
// ---------------------------------------------------------------------------
__device__ inline void async_ld_b128(unsigned lds_byte_addr, const void* sbase,
                                     unsigned byte_off) {
    asm volatile("global_load_async_to_lds_b128 %0, %1, %2 offset:0"
                 :
                 : "v"(lds_byte_addr), "v"(byte_off), "s"(sbase)
                 : "memory");
}

__device__ inline void wait_async0() {
#if defined(__has_builtin)
#if __has_builtin(__builtin_amdgcn_s_wait_asynccnt)
    __builtin_amdgcn_s_wait_asynccnt(0);
#else
    asm volatile("s_wait_asynccnt 0x0" ::: "memory");
#endif
#else
    asm volatile("s_wait_asynccnt 0x0" ::: "memory");
#endif
}

// Flat address of a __shared__ object -> raw LDS byte offset (ISA: for the
// LDS aperture, the LDS offset is the low 32 address bits).
__device__ inline unsigned lds_addr32(const void* p) {
    return (unsigned)(uintptr_t)p;
}

// ---------------------------------------------------------------------------
// Small prep kernels
// ---------------------------------------------------------------------------
__global__ __launch_bounds__(256) void cvt_f16_kernel(const float* __restrict__ in,
                                                      _Float16* __restrict__ out, int n) {
    int i = blockIdx.x * 256 + threadIdx.x;
    if (i < n) out[i] = (_Float16)in[i];
}

// bn_params: [4 branches][3 bns][4 params: g,b,m,v][C]
// aff out:   [4 branches][4: s0,b0,s12,b12][C]
__global__ __launch_bounds__(256) void fold_bn_kernel(const float* __restrict__ bnp,
                                                      float* __restrict__ aff) {
    int j = blockIdx.x, c = threadIdx.x;
    const float eps = 1e-5f;
    auto P = [&](int bni, int p) { return bnp[((j * 3 + bni) * 4 + p) * CHN + c]; };
    float s0 = P(0, 0) * rsqrtf(P(0, 3) + eps);
    float b0 = P(0, 1) - P(0, 2) * s0;
    float s1 = P(1, 0) * rsqrtf(P(1, 3) + eps);
    float b1 = P(1, 1) - P(1, 2) * s1;
    float s2 = P(2, 0) * rsqrtf(P(2, 3) + eps);
    float b2 = P(2, 1) - P(2, 2) * s2;
    aff[(j * 4 + 0) * CHN + c] = s0;
    aff[(j * 4 + 1) * CHN + c] = b0;
    aff[(j * 4 + 2) * CHN + c] = s1 * s2;
    aff[(j * 4 + 3) * CHN + c] = b1 * s2 + b2;
}

// head LIF: xs = (x >= 2.0) as fp16 0/1   [spike(x*0.5 - 1.0)]
__global__ __launch_bounds__(256) void head_lif_kernel(const float* __restrict__ x,
                                                       _Float16* __restrict__ xs) {
    size_t base = ((size_t)blockIdx.x * 256 + threadIdx.x) * 4;
    float4 v = *reinterpret_cast<const float4*>(x + base);
    xs[base + 0] = (_Float16)(v.x >= 2.0f ? 1.0f : 0.0f);
    xs[base + 1] = (_Float16)(v.y >= 2.0f ? 1.0f : 0.0f);
    xs[base + 2] = (_Float16)(v.z >= 2.0f ? 1.0f : 0.0f);
    xs[base + 3] = (_Float16)(v.w >= 2.0f ? 1.0f : 0.0f);
}

// ---------------------------------------------------------------------------
// WMMA fragment helpers (ISA 7.12.2 layouts, wave32)
//   A 16x32 f16 : lane&15 = row M; element idx -> K = (idx<8 ? idx : idx+8) + 8*(lane>>4)
//   B 32x16 f16 : lane&15 = col N; element idx -> K = (idx<8 ? idx : idx+8) + 8*(lane>>4)
//   C/D 16x16 f32: lane&15 = col N; element r  -> row M = r + 8*(lane>>4)
// ---------------------------------------------------------------------------
__device__ inline v16h frag_from_rows(const _Float16* row_base, int hi) {
    v8h lo = *reinterpret_cast<const v8h*>(row_base + 8 * hi);
    v8h hi8 = *reinterpret_cast<const v8h*>(row_base + 16 + 8 * hi);
    return __builtin_shufflevector(lo, hi8, 0, 1, 2, 3, 4, 5, 6, 7, 8, 9, 10, 11, 12, 13, 14, 15);
}

// ---------------------------------------------------------------------------
// conv1x1 as GEMM:  Out[b][o][n] = epi( sum_c Wm[o][c] * X[b][c][n] )
// M = O = 256 (2 tiles of 128), N = HW = 6400 (50 tiles of 128), K = C = 256.
// 8 waves/WG, each wave owns a 32x64 sub-tile (2x4 WMMA accumulators).
// Tiles staged by async DMA into double-buffered LDS (copy k+1 || compute k).
// EPI: 0 = affine->fp16, 1 = affine,LIF->fp16 (0/1), 2 = affine->fp32
// ---------------------------------------------------------------------------
#define TM 128
#define TN 128
#define TK 32
#define LDA 40    // halves; 80B row -> 16B-aligned vector frag loads
#define LDB 136   // halves

template <int EPI>
__global__ __launch_bounds__(256) void gemm_conv1x1_kernel(
    const _Float16* __restrict__ Wm,  // [256][256] (branch slice)
    const _Float16* __restrict__ X,   // [B][C][HW]
    void* __restrict__ Out,           // [B][O][HW] fp16 or fp32
    const float* __restrict__ scale, const float* __restrict__ bias, float lif_th) {
    __shared__ __attribute__((aligned(32))) _Float16 As[2][TM * LDA];
    __shared__ __attribute__((aligned(32))) _Float16 Bs[2][TK * LDB];

    const int bx = blockIdx.x;  // N tile
    const int by = blockIdx.y;  // M tile
    const int b  = blockIdx.z;  // batch
    const int tid = threadIdx.x;
    const int wave = tid >> 5, lane = tid & 31;
    const int ln = lane & 15, hi = lane >> 4;
    const int waveM = wave & 3, waveN = wave >> 2;

    v8f acc[2][4];
    const v8f vzero = {0.f, 0.f, 0.f, 0.f, 0.f, 0.f, 0.f, 0.f};
#pragma unroll
    for (int mi = 0; mi < 2; ++mi)
#pragma unroll
        for (int ni = 0; ni < 4; ++ni) acc[mi][ni] = vzero;

    const int ar = tid >> 1, ac = (tid & 1) * 16;  // A coop-DMA coords (row, col halves)
    const int br = tid >> 3, bc = (tid & 7) * 16;  // B coop-DMA coords

    const unsigned asb = lds_addr32(&As[0][0]);
    const unsigned bsb = lds_addr32(&Bs[0][0]);
    const unsigned aLds = (unsigned)(ar * LDA + ac) * 2u;
    const unsigned bLds = (unsigned)(br * LDB + bc) * 2u;
    const unsigned aOffBase = (unsigned)((by * TM + ar) * CHN + ac) * 2u;
    const unsigned bOffBase =
        (unsigned)(((size_t)(b * CHN + br)) * HW + (size_t)bx * TN + bc) * 2u;

    // Issue one K-slab's tiles into LDS buffer `buf` via async DMA (4x B128/thread).
    auto issue_tiles = [&](int k0, int buf) {
        const unsigned abuf = asb + (unsigned)buf * (TM * LDA * 2) + aLds;
        const unsigned ao = aOffBase + (unsigned)k0 * 2u;
        async_ld_b128(abuf, Wm, ao);
        async_ld_b128(abuf + 16, Wm, ao + 16);
        const unsigned bbuf = bsb + (unsigned)buf * (TK * LDB * 2) + bLds;
        const unsigned bo = bOffBase + (unsigned)k0 * (HW * 2u);
        async_ld_b128(bbuf, X, bo);
        async_ld_b128(bbuf + 16, X, bo + 16);
    };

    issue_tiles(0, 0);
    int cur = 0;

    for (int k0 = 0; k0 < CHN; k0 += TK) {
        wait_async0();       // own DMA into buf[cur] complete
        __syncthreads();     // everyone's DMA complete / everyone done reading buf[cur^1]

        if (k0 + TK < CHN) {
            issue_tiles(k0 + TK, cur ^ 1);  // overlap next-slab DMA with WMMAs below
            __builtin_prefetch(&X[((size_t)(b * CHN + k0 + TK + br)) * HW + bx * TN + bc], 0, 1);
        }

        const _Float16* Ac = As[cur];
        const _Float16* Bc = Bs[cur];

        v16h af[2];
#pragma unroll
        for (int mi = 0; mi < 2; ++mi)
            af[mi] = frag_from_rows(&Ac[(waveM * 32 + mi * 16 + ln) * LDA], hi);

        v16h bf[4];
#pragma unroll
        for (int ni = 0; ni < 4; ++ni) {
            const int n = waveN * 64 + ni * 16 + ln;
#pragma unroll
            for (int idx = 0; idx < 16; ++idx)
                bf[ni][idx] = Bc[((idx & 7) + 8 * hi + 16 * (idx >> 3)) * LDB + n];
        }

#pragma unroll
        for (int mi = 0; mi < 2; ++mi)
#pragma unroll
            for (int ni = 0; ni < 4; ++ni)
                acc[mi][ni] = __builtin_amdgcn_wmma_f32_16x16x32_f16(
                    false, af[mi], false, bf[ni], (short)0, acc[mi][ni], false, false);

        cur ^= 1;
    }

    // Epilogue: per-channel affine (+ optional LIF), store.
#pragma unroll
    for (int mi = 0; mi < 2; ++mi) {
        const int mbase = by * TM + waveM * 32 + mi * 16;
#pragma unroll
        for (int ni = 0; ni < 4; ++ni) {
            const int n = bx * TN + waveN * 64 + ni * 16 + ln;
#pragma unroll
            for (int r = 0; r < 8; ++r) {
                const int o = mbase + r + 8 * hi;
                float val = scale[o] * acc[mi][ni][r] + bias[o];
                const size_t oidx = ((size_t)b * CHN + o) * HW + n;
                if (EPI == 2) {
                    reinterpret_cast<float*>(Out)[oidx] = val;
                } else {
                    if (EPI == 1) val = (0.5f * val - lif_th >= 0.0f) ? 1.0f : 0.0f;
                    reinterpret_cast<_Float16*>(Out)[oidx] = (_Float16)val;
                }
            }
        }
    }
}

// ---------------------------------------------------------------------------
// depthwise 3x3, SAME padding, fp16 in/out
// ---------------------------------------------------------------------------
__global__ __launch_bounds__(256) void dwconv3x3_kernel(const _Float16* __restrict__ in,
                                                        const _Float16* __restrict__ wdw,
                                                        _Float16* __restrict__ out) {
    const size_t i = (size_t)blockIdx.x * 256 + threadIdx.x;
    const int w = (int)(i % WW);
    const int h = (int)((i / WW) % HH);
    const size_t bc = i / HW;
    const int c = (int)(bc % CHN);
    const _Float16* wp = wdw + c * 9;
    const _Float16* ip = in + bc * HW;
    float s = 0.0f;
#pragma unroll
    for (int dy = -1; dy <= 1; ++dy) {
        const int hh = h + dy;
        if (hh < 0 || hh >= HH) continue;
#pragma unroll
        for (int dx = -1; dx <= 1; ++dx) {
            const int ww2 = w + dx;
            if (ww2 < 0 || ww2 >= WW) continue;
            s += (float)wp[(dy + 1) * 3 + (dx + 1)] * (float)ip[hh * WW + ww2];
        }
    }
    out[i] = (_Float16)s;
}

// ---------------------------------------------------------------------------
// Attention step 1: kv[b,c,w,u] = sum_h K[b,c,h,w] * V[b,c,h,u]
// One WG (4 waves) per (b,c). Operands staged column-major in LDS with the
// contraction dim padded 80->96 (zeros) so K-loops are 3 clean WMMA steps.
// ---------------------------------------------------------------------------
#define KPAD 96

__global__ __launch_bounds__(128) void attn_kv_kernel(const _Float16* __restrict__ K_,
                                                      const _Float16* __restrict__ V_,
                                                      _Float16* __restrict__ KV) {
    const int bcid = blockIdx.x;
    const _Float16* kp = K_ + (size_t)bcid * HW;
    const _Float16* vp = V_ + (size_t)bcid * HW;
    __shared__ __attribute__((aligned(32))) _Float16 Ksh[WW * KPAD];  // [w][h(pad96)]
    __shared__ __attribute__((aligned(32))) _Float16 Vsh[WW * KPAD];  // [u][h(pad96)]
    const int tid = threadIdx.x;
    for (int i = tid; i < HW; i += 128) {
        const int h = i / WW, w = i % WW;
        Ksh[w * KPAD + h] = kp[i];
        Vsh[w * KPAD + h] = vp[i];
    }
    for (int i = tid; i < WW * (KPAD - HH); i += 128) {
        const int col = i / (KPAD - HH), kk = HH + (i % (KPAD - HH));
        Ksh[col * KPAD + kk] = (_Float16)0.0f;
        Vsh[col * KPAD + kk] = (_Float16)0.0f;
    }
    __syncthreads();

    const int wave = tid >> 5, lane = tid & 31, ln = lane & 15, hi = lane >> 4;
    _Float16* outp = KV + (size_t)bcid * HW;
    const v8f vzero = {0.f, 0.f, 0.f, 0.f, 0.f, 0.f, 0.f, 0.f};

    for (int t = wave; t < 25; t += 4) {
        const int tm = t / 5, tn = t % 5;
        v8f acc = vzero;
#pragma unroll
        for (int ks = 0; ks < KPAD; ks += 32) {
            v16h a = frag_from_rows(&Ksh[(tm * 16 + ln) * KPAD + ks], hi);
            v16h bfr = frag_from_rows(&Vsh[(tn * 16 + ln) * KPAD + ks], hi);
            acc = __builtin_amdgcn_wmma_f32_16x16x32_f16(false, a, false, bfr, (short)0, acc,
                                                         false, false);
        }
#pragma unroll
        for (int r = 0; r < 8; ++r) {
            const int m = tm * 16 + r + 8 * hi;
            const int n = tn * 16 + ln;
            outp[m * WW + n] = (_Float16)acc[r];
        }
    }
}

// ---------------------------------------------------------------------------
// Attention step 2: s[b,c,h,u] = LIF_0.5( 0.125 * sum_w q[b,c,h,w]*kv[b,c,w,u] )
// ---------------------------------------------------------------------------
__global__ __launch_bounds__(128) void attn_qk_kernel(const _Float16* __restrict__ Q_,
                                                      const _Float16* __restrict__ KV,
                                                      _Float16* __restrict__ S) {
    const int bcid = blockIdx.x;
    const _Float16* qp = Q_ + (size_t)bcid * HW;
    const _Float16* kvp = KV + (size_t)bcid * HW;
    __shared__ __attribute__((aligned(32))) _Float16 Qsh[HH * KPAD];   // [h][w(pad96)]
    __shared__ __attribute__((aligned(32))) _Float16 KVsh[WW * KPAD];  // [u][w(pad96)]
    const int tid = threadIdx.x;
    for (int i = tid; i < HW; i += 128) {
        const int r = i / WW, c = i % WW;
        Qsh[r * KPAD + c] = qp[i];    // A[m=h][k=w]
        KVsh[c * KPAD + r] = kvp[i];  // B[k=w][n=u] stored transposed
    }
    for (int i = tid; i < WW * (KPAD - HH); i += 128) {
        const int col = i / (KPAD - HH), kk = HH + (i % (KPAD - HH));
        Qsh[col * KPAD + kk] = (_Float16)0.0f;
        KVsh[col * KPAD + kk] = (_Float16)0.0f;
    }
    __syncthreads();

    const int wave = tid >> 5, lane = tid & 31, ln = lane & 15, hi = lane >> 4;
    _Float16* outp = S + (size_t)bcid * HW;
    const v8f vzero = {0.f, 0.f, 0.f, 0.f, 0.f, 0.f, 0.f, 0.f};

    for (int t = wave; t < 25; t += 4) {
        const int tm = t / 5, tn = t % 5;
        v8f acc = vzero;
#pragma unroll
        for (int ks = 0; ks < KPAD; ks += 32) {
            v16h a = frag_from_rows(&Qsh[(tm * 16 + ln) * KPAD + ks], hi);
            v16h bfr = frag_from_rows(&KVsh[(tn * 16 + ln) * KPAD + ks], hi);
            acc = __builtin_amdgcn_wmma_f32_16x16x32_f16(false, a, false, bfr, (short)0, acc,
                                                         false, false);
        }
#pragma unroll
        for (int r = 0; r < 8; ++r) {
            const int m = tm * 16 + r + 8 * hi;
            const int n = tn * 16 + ln;
            const float attn = acc[r] * 0.125f;
            // lif(attn, 0.5): spike(attn*0.5 - 0.5) -> attn >= 1.0
            outp[m * WW + n] = (_Float16)((0.5f * attn - 0.5f >= 0.0f) ? 1.0f : 0.0f);
        }
    }
}

// ---------------------------------------------------------------------------
// Host-side orchestration
// ---------------------------------------------------------------------------
extern "C" void kernel_launch(void* const* d_in, const int* in_sizes, int n_in,
                              void* d_out, int out_size, void* d_ws, size_t ws_size,
                              hipStream_t stream) {
    const float* xf  = (const float*)d_in[0];  // [16,256,80,80]
    const float* w1f = (const float*)d_in[1];  // [4,256,256]
    const float* dwf = (const float*)d_in[2];  // [4,256,3,3]
    const float* w2f = (const float*)d_in[3];  // [4,256,256]
    const float* bnp = (const float*)d_in[4];  // [4,3,4,256]

    char* ws = (char*)d_ws;
    const size_t HB = NELEM * sizeof(_Float16);  // 50 MiB per fp16 buffer
    _Float16* xs = (_Float16*)(ws + 0 * HB);     // spikes; reused for s
    _Float16* t1 = (_Float16*)(ws + 1 * HB);
    _Float16* t2 = (_Float16*)(ws + 2 * HB);
    _Float16* qb = (_Float16*)(ws + 3 * HB);
    _Float16* kb = (_Float16*)(ws + 4 * HB);
    _Float16* vb = (_Float16*)(ws + 5 * HB);
    _Float16* kv = (_Float16*)(ws + 6 * HB);
    _Float16* w1h = (_Float16*)(ws + 7 * HB);
    _Float16* w2h = w1h + 4 * CHN * CHN;
    _Float16* dwh = w2h + 4 * CHN * CHN;
    float* aff = (float*)(dwh + 4 * CHN * 9);  // [4][4][256] floats

    // Weight conversion + BN folding
    cvt_f16_kernel<<<(4 * CHN * CHN + 255) / 256, 256, 0, stream>>>(w1f, w1h, 4 * CHN * CHN);
    cvt_f16_kernel<<<(4 * CHN * CHN + 255) / 256, 256, 0, stream>>>(w2f, w2h, 4 * CHN * CHN);
    cvt_f16_kernel<<<(4 * CHN * 9 + 255) / 256, 256, 0, stream>>>(dwf, dwh, 4 * CHN * 9);
    fold_bn_kernel<<<4, 256, 0, stream>>>(bnp, aff);

    // head LIF
    head_lif_kernel<<<(unsigned)(NELEM / 1024), 256, 0, stream>>>(xf, xs);

    const dim3 gg(HW / TN, CHN / TM, BATCH);  // (50, 2, 16)
    const unsigned dwg = (unsigned)(NELEM / 256);

    _Float16* branch_out[3] = {qb, kb, vb};
    for (int j = 0; j < 3; ++j) {
        const float* s0  = aff + (j * 4 + 0) * CHN;
        const float* b0  = aff + (j * 4 + 1) * CHN;
        const float* s12 = aff + (j * 4 + 2) * CHN;
        const float* b12 = aff + (j * 4 + 3) * CHN;
        gemm_conv1x1_kernel<0><<<gg, 256, 0, stream>>>(w1h + j * CHN * CHN, xs, t1, s0, b0, 0.f);
        dwconv3x3_kernel<<<dwg, 256, 0, stream>>>(t1, dwh + j * CHN * 9, t2);
        if (j == 2)  // v branch: fuse v_lif (threshold 1.0) -> binary spikes
            gemm_conv1x1_kernel<1><<<gg, 256, 0, stream>>>(w2h + j * CHN * CHN, t2,
                                                           branch_out[j], s12, b12, 1.0f);
        else
            gemm_conv1x1_kernel<0><<<gg, 256, 0, stream>>>(w2h + j * CHN * CHN, t2,
                                                           branch_out[j], s12, b12, 0.f);
    }

    // Attention: kv = K^T V, then s = LIF(Q kv * 0.125)
    attn_kv_kernel<<<BATCH * CHN, 128, 0, stream>>>(kb, vb, kv);
    attn_qk_kernel<<<BATCH * CHN, 128, 0, stream>>>(qb, kv, xs);  // xs reused as s

    // proj branch -> fp32 output
    {
        const int j = 3;
        const float* s0  = aff + (j * 4 + 0) * CHN;
        const float* b0  = aff + (j * 4 + 1) * CHN;
        const float* s12 = aff + (j * 4 + 2) * CHN;
        const float* b12 = aff + (j * 4 + 3) * CHN;
        gemm_conv1x1_kernel<0><<<gg, 256, 0, stream>>>(w1h + j * CHN * CHN, xs, t1, s0, b0, 0.f);
        dwconv3x3_kernel<<<dwg, 256, 0, stream>>>(t1, dwh + j * CHN * 9, t2);
        gemm_conv1x1_kernel<2><<<gg, 256, 0, stream>>>(w2h + j * CHN * CHN, t2, d_out, s12, b12,
                                                       0.f);
    }
    (void)in_sizes; (void)n_in; (void)out_size; (void)ws_size;
}
